// TaxiFairQNetwork_78958678770187
// MI455X (gfx1250) — matrix operand order repaired and checked by reading, not compiled, not run on hardware
//
#include <hip/hip_runtime.h>

typedef _Float16 v16h __attribute__((ext_vector_type(16)));
typedef _Float16 v8h  __attribute__((ext_vector_type(8)));
typedef float    v8f  __attribute__((ext_vector_type(8)));

namespace {
constexpr int   kBatch = 256;
constexpr int   kC     = 4096;
constexpr int   kF     = 64;
constexpr int   kG     = 512;
constexpr int   kH1    = 128;
constexpr int   kH2    = 128;
constexpr int   kHB    = 32;
constexpr float kNeg   = -1000000000.0f;
constexpr int   kRows  = kBatch * kC;     // 1048576
constexpr int   kNSeg  = kBatch * kG;     // 131072
constexpr int   MTILE  = 128;             // rows per block (8 waves x 16 rows)
constexpr int   SA_STR = kF + 8;          // 72 halves (144B, 16B aligned)
constexpr int   SH_STR = kH1 + 8;         // 136 halves (272B, 16B aligned)
constexpr int   W0T_STR = kF;             // f16 [n][k], k contiguous
constexpr int   W1T_STR = kH1;
}

__device__ __forceinline__ v16h cat8(v8h lo, v8h hi) {
  return __builtin_shufflevector(lo, hi, 0,1,2,3,4,5,6,7,8,9,10,11,12,13,14,15);
}

// ---------------------------------------------------------------------------
// Kernel 1: transpose+convert weights to f16 [N][K]; init segment accumulators
// ---------------------------------------------------------------------------
__global__ __launch_bounds__(256) void prep_kernel(
    const float* __restrict__ W0, const float* __restrict__ W1,
    _Float16* __restrict__ w0t, _Float16* __restrict__ w1t,
    float* __restrict__ cnt, float* __restrict__ ssum,
    unsigned* __restrict__ smaxkey)
{
  int idx = blockIdx.x * 256 + threadIdx.x;
  if (idx < kF * kH1) {               // W0: (64,128) row-major -> w0t[n][k]
    int k = idx >> 7, n = idx & (kH1 - 1);
    w0t[n * W0T_STR + k] = (_Float16)W0[idx];
  }
  if (idx < kH1 * kH2) {              // W1: (128,128) -> w1t[n][k]
    int k = idx >> 7, n = idx & (kH2 - 1);
    w1t[n * W1T_STR + k] = (_Float16)W1[idx];
  }
  if (idx < kNSeg) {
    cnt[idx]  = 0.0f;
    ssum[idx] = 0.0f;
    smaxkey[idx] = ~__float_as_uint(kNeg);   // key(NEG): NEG<0 -> key = ~bits
  }
}

// ---------------------------------------------------------------------------
// Kernel 2: fused MLP (WMMA f16) -> raw, mask-apply, segment atomics
// ---------------------------------------------------------------------------
__global__ __launch_bounds__(256) void mlp_raw_kernel(
    const float* __restrict__ x,
    const _Float16* __restrict__ w0t, const _Float16* __restrict__ w1t,
    const float* __restrict__ b0, const float* __restrict__ b1,
    const float* __restrict__ W2, const float* __restrict__ b2,
    float* __restrict__ raw_ws,
    float* __restrict__ cnt, float* __restrict__ ssum,
    unsigned* __restrict__ smaxkey)
{
  __shared__ __align__(16) _Float16 sA[MTILE * SA_STR];   // feats tile, f16
  __shared__ __align__(16) _Float16 sH[MTILE * SH_STR];   // relu(h1), f16
  __shared__ float sB0[kH1], sB1[kH2], sW2v[kH2], sRaw[MTILE];

  const int tid = threadIdx.x;
  const int rowBase = blockIdx.x * MTILE;

  // Stage feats tile -> LDS f16 (coalesced 64-float runs per row)
  for (int idx = tid; idx < MTILE * kF; idx += 256) {
    int r = idx >> 6, c = idx & 63;
    sA[r * SA_STR + c] = (_Float16)x[(size_t)(rowBase + r) * 66 + c];
  }
  if (tid < kH1) sB0[tid] = b0[tid];
  if (tid < kH2) { sB1[tid] = b1[tid]; sW2v[tid] = W2[tid]; }
  __syncthreads();

  const int wave  = tid >> 5;
  const int lane  = tid & 31;
  const int lhalf = lane >> 4;     // 0: lanes 0-15, 1: lanes 16-31
  const int l16   = lane & 15;
  const int m0    = wave * 16;     // this wave's 16 rows within the tile

  // ---- layer 1: h1 = relu(feats @ W0 + b0), 16 rows x 128 cols per wave ----
  for (int nt = 0; nt < 8; ++nt) {
    const float bv = sB0[nt * 16 + l16];
    v8f c = {bv, bv, bv, bv, bv, bv, bv, bv};
#pragma unroll
    for (int ks = 0; ks < 2; ++ks) {
      const int kb = ks * 32;
      // A fragment: 16-bit A 16x32 layout
      const _Float16* ap = &sA[(m0 + l16) * SA_STR + kb + lhalf * 8];
      v16h a = cat8(*(const v8h*)ap, *(const v8h*)(ap + 16));
      // B fragment: 16-bit B 32x16 layout from transposed f16 weights
      const _Float16* bp = &w0t[(nt * 16 + l16) * W0T_STR + kb + lhalf * 8];
      v16h bm = cat8(*(const v8h*)bp, *(const v8h*)(bp + 16));
      c = __builtin_amdgcn_wmma_f32_16x16x32_f16(false, a, false, bm,
                                                 (short)0, c, false, false);
    }
#pragma unroll
    for (int r = 0; r < 8; ++r) {
      float v = fmaxf(c[r], 0.0f);  // relu
      sH[(m0 + r + 8 * lhalf) * SH_STR + nt * 16 + l16] = (_Float16)v;
    }
  }
  __syncthreads();

  // ---- layer 2: h2 = relu(h1 @ W1 + b1); fold W2 dot in-register ----
  float acc[8] = {0.f, 0.f, 0.f, 0.f, 0.f, 0.f, 0.f, 0.f};
  for (int nt = 0; nt < 8; ++nt) {
    const float bv = sB1[nt * 16 + l16];
    v8f c = {bv, bv, bv, bv, bv, bv, bv, bv};
#pragma unroll
    for (int ks = 0; ks < 4; ++ks) {
      const int kb = ks * 32;
      const _Float16* ap = &sH[(m0 + l16) * SH_STR + kb + lhalf * 8];
      v16h a = cat8(*(const v8h*)ap, *(const v8h*)(ap + 16));
      const _Float16* bp = &w1t[(nt * 16 + l16) * W1T_STR + kb + lhalf * 8];
      v16h bm = cat8(*(const v8h*)bp, *(const v8h*)(bp + 16));
      c = __builtin_amdgcn_wmma_f32_16x16x32_f16(false, a, false, bm,
                                                 (short)0, c, false, false);
    }
    const float w2v = sW2v[nt * 16 + l16];
#pragma unroll
    for (int r = 0; r < 8; ++r)
      acc[r] += fmaxf(c[r], 0.0f) * w2v;
  }
  // reduce over the 16-lane N group (xor masks < 16 stay within group)
#pragma unroll
  for (int off = 1; off < 16; off <<= 1)
#pragma unroll
    for (int r = 0; r < 8; ++r)
      acc[r] += __shfl_xor(acc[r], off, 32);
  if (l16 == 0) {
    const float bb = b2[0];
#pragma unroll
    for (int r = 0; r < 8; ++r)
      sRaw[m0 + 8 * lhalf + r] = acc[r] + bb;
  }
  __syncthreads();

  // ---- epilogue: mask, write raw, per-segment atomics ----
  if (tid < MTILE) {
    const size_t gr = (size_t)rowBase + tid;
    const float gidf = x[gr * 66 + 64];
    const float mask = x[gr * 66 + 65];
    const int   gid  = (int)gidf;
    const float raw  = sRaw[tid];
    const bool  mpos = mask > 0.0f;
    const float rawm = mpos ? raw : kNeg;
    raw_ws[gr] = rawm;
    if (mpos && gid >= 0) {
      const int b   = (int)(gr >> 12);    // C = 4096
      const int seg = b * kG + gid;
      atomicAdd(&cnt[seg], 1.0f);
      atomicAdd(&ssum[seg], rawm);
      unsigned u = __float_as_uint(rawm);
      unsigned key = (u & 0x80000000u) ? ~u : (u | 0x80000000u);
      atomicMax(&smaxkey[seg], key);
    }
  }
}

// ---------------------------------------------------------------------------
// Kernel 3: per-segment stats + tiny bias MLP (3 -> 32 -> 1), VALU only
// ---------------------------------------------------------------------------
__global__ __launch_bounds__(256) void stats_kernel(
    const float* __restrict__ cnt, const float* __restrict__ ssum,
    const unsigned* __restrict__ smaxkey,
    const float* __restrict__ Wb0, const float* __restrict__ bb0,
    const float* __restrict__ Wb1, const float* __restrict__ bb1,
    float* __restrict__ meanA, float* __restrict__ smaxA,
    float* __restrict__ biasA, float* __restrict__ multiA)
{
  __shared__ float sWb0[3 * kHB], sbb0[kHB], sWb1[kHB];
  const int tid = threadIdx.x;
  if (tid < 3 * kHB) sWb0[tid] = Wb0[tid];
  if (tid < kHB) { sbb0[tid] = bb0[tid]; sWb1[tid] = Wb1[tid]; }
  __syncthreads();

  const int s = blockIdx.x * 256 + tid;
  if (s >= kNSeg) return;

  const float c = cnt[s];
  const unsigned key = smaxkey[s];
  const unsigned u = (key & 0x80000000u) ? (key & 0x7fffffffu) : ~key;
  const float maxf = __uint_as_float(u);
  const float sm   = (c > 0.0f) ? maxf : 0.0f;
  const float mean = ssum[s] / fmaxf(c, 1.0f);
  const bool  multi = c > 1.0f;
  const float s0 = multi ? sm   : 0.0f;
  const float s1 = multi ? mean : 0.0f;
  const float s2 = multi ? c    : 0.0f;

  float bias = bb1[0];
#pragma unroll
  for (int j = 0; j < kHB; ++j) {
    float h = s0 * sWb0[j] + s1 * sWb0[kHB + j] + s2 * sWb0[2 * kHB + j] + sbb0[j];
    bias += fmaxf(h, 0.0f) * sWb1[j];
  }
  meanA[s] = mean;
  smaxA[s] = sm;
  biasA[s] = bias;
  multiA[s] = multi ? 1.0f : 0.0f;
}

// ---------------------------------------------------------------------------
// Kernel 4: gather + fair formula, write (fair, mask)
// ---------------------------------------------------------------------------
__global__ __launch_bounds__(256) void final_kernel(
    const float* __restrict__ x, const float* __restrict__ raw_ws,
    const float* __restrict__ meanA, const float* __restrict__ smaxA,
    const float* __restrict__ biasA, const float* __restrict__ multiA,
    float* __restrict__ out)
{
  const size_t i = (size_t)blockIdx.x * 256 + threadIdx.x;
  const float gidf = x[i * 66 + 64];
  const float mask = x[i * 66 + 65];
  const int   gid  = (int)gidf;
  const float rawm = raw_ws[i];
  float fair = rawm;
  if (mask > 0.0f && gid >= 0) {
    const int b = (int)(i >> 12);
    const int seg = b * kG + gid;
    if (multiA[seg] > 0.5f)
      fair = 1.5f * rawm - meanA[seg] + 0.5f * smaxA[seg] + biasA[seg];
  }
  fair = (mask > 0.0f) ? fair : kNeg;
  out[i] = fair;
  out[(size_t)kRows + i] = mask;
}

// ---------------------------------------------------------------------------
extern "C" void kernel_launch(void* const* d_in, const int* in_sizes, int n_in,
                              void* d_out, int out_size, void* d_ws, size_t ws_size,
                              hipStream_t stream)
{
  const float* x   = (const float*)d_in[0];
  const float* W0  = (const float*)d_in[1];
  const float* b0  = (const float*)d_in[2];
  const float* W1  = (const float*)d_in[3];
  const float* b1  = (const float*)d_in[4];
  const float* W2  = (const float*)d_in[5];
  const float* b2  = (const float*)d_in[6];
  const float* Wb0 = (const float*)d_in[7];
  const float* bb0 = (const float*)d_in[8];
  const float* Wb1 = (const float*)d_in[9];
  const float* bb1 = (const float*)d_in[10];
  float* out = (float*)d_out;

  // workspace partition (all offsets 16B aligned)
  char* p = (char*)d_ws;
  _Float16* w0t   = (_Float16*)p;  p += (size_t)kF  * kH1 * sizeof(_Float16); // 16 KB
  _Float16* w1t   = (_Float16*)p;  p += (size_t)kH1 * kH2 * sizeof(_Float16); // 32 KB
  float* raw_ws   = (float*)p;     p += (size_t)kRows * sizeof(float);        // 4 MB
  float* cnt      = (float*)p;     p += (size_t)kNSeg * sizeof(float);
  float* ssum     = (float*)p;     p += (size_t)kNSeg * sizeof(float);
  unsigned* smaxk = (unsigned*)p;  p += (size_t)kNSeg * sizeof(unsigned);
  float* meanA    = (float*)p;     p += (size_t)kNSeg * sizeof(float);
  float* smaxA    = (float*)p;     p += (size_t)kNSeg * sizeof(float);
  float* biasA    = (float*)p;     p += (size_t)kNSeg * sizeof(float);
  float* multiA   = (float*)p;

  prep_kernel<<<kNSeg / 256, 256, 0, stream>>>(W0, W1, w0t, w1t, cnt, ssum, smaxk);
  mlp_raw_kernel<<<kRows / MTILE, 256, 0, stream>>>(
      x, w0t, w1t, b0, b1, W2, b2, raw_ws, cnt, ssum, smaxk);
  stats_kernel<<<kNSeg / 256, 256, 0, stream>>>(
      cnt, ssum, smaxk, Wb0, bb0, Wb1, bb1, meanA, smaxA, biasA, multiA);
  final_kernel<<<kRows / 256, 256, 0, stream>>>(
      x, raw_ws, meanA, smaxA, biasA, multiA, out);
}
